// AttentionAggregationModule_75402445849148
// MI455X (gfx1250) — matrix operand order repaired and checked by reading, not compiled, not run on hardware
//
#include <hip/hip_runtime.h>

// ---------------- problem constants ----------------
#define NB     4
#define CIN    256
#define COUT   64
#define CQ     8
#define NSP    65536              // pixels per image (256*256)
#define NPIX   (NB * NSP)         // 262144
#define BN_EPS 1e-5f
#define EPS_ATT 1e-6f

// ---------------- workspace layout (float offsets) ----------------
#define WS_X        0                       // x / feat (in place), NPIX*COUT floats
#define WS_KT_F     (NPIX * COUT)           // 16,777,216 : Kt as _Float16 [pixel][16] -> 2,097,152 float slots
#define WS_STATS    (WS_KT_F + (NPIX * 16 / 2))
// stats region (relative float offsets)
#define ST_GSUM     0
#define ST_GSUMSQ   64
#define ST_FEATSUM  128                     // [B][64]
#define ST_KSUM     384                     // [B][8] (padded)
#define ST_KFT      416                     // [B][64][16]  (KF transposed, q padded to 16)
#define ST_ZERO_N   4512                    // everything above is atomically accumulated
#define ST_SCALE    4512
#define ST_SHIFT    4576
#define ST_KSEPS    4640                    // [B][8] padded
#define ST_VSUM     4672                    // [B][64]
#define ST_MAT      4928                    // [B][8][64]

typedef _Float16 v16h __attribute__((ext_vector_type(16)));
typedef _Float16 v8h  __attribute__((ext_vector_type(8)));
typedef float    v8f  __attribute__((ext_vector_type(8)));

static __device__ __forceinline__ v8f wmma_f16(v16h a, v16h b, v8f c) {
    return __builtin_amdgcn_wmma_f32_16x16x32_f16(false, a, false, b, (short)0, c, false, false);
}

// ================= K0: zero atomic accumulators =================
__global__ void k0_zero(float* stats) {
    int i = blockIdx.x * 256 + threadIdx.x;
    if (i < ST_ZERO_N) stats[i] = 0.0f;
}

// ================= K1: fused 1x1 conv (WMMA) + BN stats =================
// GEMM: x[pix, o] = sum_c W[o,c] * fcat[c, pix].  A = W (16o x 32c, f16, from LDS),
// B = fcat (32c x 16pix, f16, from global), D = 16o x 16pix, f32 accum.
__global__ __launch_bounds__(256) void k1_conv(
    const float* __restrict__ s5, const float* __restrict__ s4,
    const float* __restrict__ s3, const float* __restrict__ s2,
    const float* __restrict__ conv_w,
    float* __restrict__ x, float* __restrict__ gsum, float* __restrict__ gsumsq)
{
    __shared__ __attribute__((aligned(16))) _Float16 wh[64 * 264]; // padded stride (16B) kills DS bank conflicts

    const int tid  = threadIdx.x;
    const int wave = tid >> 5;
    const int lane = tid & 31;
    const int lx   = lane & 15;       // column / sub-row index
    const int hi   = lane >> 4;       // half-wave select

    // stage weights into LDS as f16 (A-matrix source)
    for (int i = tid; i < 64 * 256; i += 256) {
        int o = i >> 8, c = i & 255;
        wh[o * 264 + c] = (_Float16)conv_w[i];
    }
    __syncthreads();

    float ls[32], lsq[32];
    #pragma unroll
    for (int i = 0; i < 32; ++i) { ls[i] = 0.0f; lsq[i] = 0.0f; }

    const int tiles_total = NPIX / 16;      // 16-pixel tiles
    const int stride = gridDim.x * 8;

    for (int tile = blockIdx.x * 8 + wave; tile < tiles_total; tile += stride) {
        const int p0 = tile << 4;
        const int b  = p0 >> 16;
        const int n0 = p0 & (NSP - 1);

        // prefetch next tile's first source line (global_prefetch_b8 path)
        {
            int nt = tile + stride;
            if (nt < tiles_total) {
                int np0 = nt << 4;
                __builtin_prefetch(s5 + (((size_t)(np0 >> 16) * 64) << 16) + (np0 & (NSP - 1)), 0, 1);
            }
        }

        v8f acc[4];
        #pragma unroll
        for (int m = 0; m < 4; ++m)
            #pragma unroll
            for (int e = 0; e < 8; ++e) acc[m][e] = 0.0f;

        for (int kb = 0; kb < 8; ++kb) {           // K blocks of 32 input channels
            const float* src = (kb < 2) ? s5 : (kb < 4) ? s4 : (kb < 6) ? s3 : s2;
            const int cbase = (kb & 1) * 32;       // channel-in-tensor base of this 32-block
            // ---- B fragment: rows = channels, cols = pixels ----
            const float* bp = src + (((size_t)(b * 64 + cbase + hi * 16)) << 16) + n0 + lx;
            v16h bfrag;
            #pragma unroll
            for (int v = 0; v < 8; ++v) {
                float f0 = bp[(size_t)(2 * v)     << 16];
                float f1 = bp[(size_t)(2 * v + 1) << 16];
                bfrag[2 * v]     = (_Float16)f0;
                bfrag[2 * v + 1] = (_Float16)f1;
            }
            // ---- A fragments (4 o-tiles), reuse B across them ----
            const int acol = kb * 32 + hi * 16;
            #pragma unroll
            for (int m = 0; m < 4; ++m) {
                const _Float16* ap = &wh[(m * 16 + lx) * 264 + acol];
                v8h alo = *(const v8h*)ap;
                v8h ahi = *(const v8h*)(ap + 8);
                v16h afrag;
                #pragma unroll
                for (int i = 0; i < 8; ++i) { afrag[i] = alo[i]; afrag[8 + i] = ahi[i]; }
                acc[m] = wmma_f16(afrag, bfrag, acc[m]);
            }
        }

        // ---- store D + local BN-stat accumulation ----
        float* xp = x + (((size_t)b) << 22) + n0 + lx;
        #pragma unroll
        for (int m = 0; m < 4; ++m)
            #pragma unroll
            for (int r = 0; r < 8; ++r) {
                int o = m * 16 + r + hi * 8;
                float v = acc[m][r];
                xp[(size_t)o << 16] = v;
                ls[m * 8 + r]  += v;
                lsq[m * 8 + r] += v * v;
            }
    }

    // reduce stats over the 16-lane groups, then global atomics
    #pragma unroll
    for (int i = 0; i < 32; ++i) {
        float s = ls[i], q = lsq[i];
        for (int m = 8; m >= 1; m >>= 1) {
            s += __shfl_xor(s, m, 16);
            q += __shfl_xor(q, m, 16);
        }
        if (lx == 0) {
            int o = (i >> 3) * 16 + (i & 7) + hi * 8;
            atomicAdd(&gsum[o], s);
            atomicAdd(&gsumsq[o], q);
        }
    }
}

// ================= K1b: fold BN into scale/shift =================
__global__ void k1b_bn(const float* __restrict__ gsum, const float* __restrict__ gsumsq,
                       const float* __restrict__ bn_w, const float* __restrict__ bn_b,
                       float* __restrict__ scale, float* __restrict__ shift)
{
    int o = threadIdx.x;
    const float inv_cnt = 1.0f / (float)NPIX;
    float mean = gsum[o] * inv_cnt;
    float var  = gsumsq[o] * inv_cnt - mean * mean;
    float sc   = bn_w[o] * rsqrtf(var + BN_EPS);
    scale[o] = sc;
    shift[o] = bn_b[o] - mean * sc;
}

// ================= K2: Mish (in place), K projection, featsum/Ksum, Kt(f16) =================
__global__ __launch_bounds__(256) void k2_feat(
    float* __restrict__ xf,                 // in: x, out: feat (in place)
    const float* __restrict__ scale, const float* __restrict__ shift,
    const float* __restrict__ k_w, const float* __restrict__ k_b,
    _Float16* __restrict__ kt,
    float* __restrict__ featsum, float* __restrict__ ksum)
{
    __shared__ float kwl[CQ * COUT];
    __shared__ float scl[COUT], shf[COUT];
    __shared__ float red_fs[COUT];
    __shared__ float red_ks[CQ];

    const int tid = threadIdx.x;
    for (int i = tid; i < CQ * COUT; i += 256) kwl[i] = k_w[i];
    if (tid < COUT) { scl[tid] = scale[tid]; shf[tid] = shift[tid]; red_fs[tid] = 0.0f; }
    if (tid < CQ) red_ks[tid] = 0.0f;
    __syncthreads();

    const int p = blockIdx.x * 256 + tid;
    const int b = p >> 16;
    const int n = p & (NSP - 1);
    const size_t base = (((size_t)b) << 22) + n;

    float feat[COUT];
    #pragma unroll 8
    for (int o = 0; o < COUT; ++o) {
        float xv = xf[base + ((size_t)o << 16)];
        float f  = xv * scl[o] + shf[o];
        float sp = (f > 20.0f) ? f : log1pf(__expf(f));    // softplus
        float fe = f * tanhf(sp);                          // Mish
        feat[o] = fe;
        xf[base + ((size_t)o << 16)] = fe;                 // feat in place
        atomicAdd(&red_fs[o], fe);
    }

    // K = l2norm(k_w @ feat + k_b)
    float kk[CQ];
    float s2 = 0.0f;
    #pragma unroll
    for (int q = 0; q < CQ; ++q) {
        float a = k_b[q];
        #pragma unroll 8
        for (int o = 0; o < COUT; ++o) a += kwl[q * COUT + o] * feat[o];
        kk[q] = a; s2 += a * a;
    }
    float rn = 1.0f / sqrtf(s2);
    _Float16* kp = kt + (size_t)p * 16;
    #pragma unroll
    for (int q = 0; q < CQ; ++q) {
        kk[q] *= rn;
        kp[q] = (_Float16)kk[q];
        kp[q + 8] = (_Float16)0.0f;       // zero-pad q=8..15 for WMMA
        atomicAdd(&red_ks[q], kk[q]);
    }

    __syncthreads();
    if (tid < COUT) atomicAdd(&featsum[b * COUT + tid], red_fs[tid]);
    if (tid < CQ)   atomicAdd(&ksum[b * CQ + tid], red_ks[tid]);
}

// ================= K2b: KF^T = feat @ Kt  (split-K WMMA GEMM) =================
// A = feat (16o x 32pix f16), B = Kt (32pix x 16q f16), accumulate f32, atomic-add partials.
__global__ __launch_bounds__(256) void k2b_kf(
    const float* __restrict__ feat, const _Float16* __restrict__ kt,
    float* __restrict__ kft)
{
    const int tid  = threadIdx.x;
    const int wave = tid >> 5;
    const int lane = tid & 31;
    const int lx   = lane & 15;
    const int hi   = lane >> 4;

    const int w  = blockIdx.x * 8 + wave;   // 1024 waves total (grid=128)
    const int b  = w >> 8;                  // 256 waves per batch
    const int wb = w & 255;

    v8f acc[4];
    #pragma unroll
    for (int m = 0; m < 4; ++m)
        #pragma unroll
        for (int e = 0; e < 8; ++e) acc[m][e] = 0.0f;

    for (int i = 0; i < 8; ++i) {           // 8 chunks of 32 pixels per wave
        const int n0 = (wb + (i << 8)) << 5;
        const size_t p0 = (((size_t)b) << 16) + n0;
        __builtin_prefetch(kt + (p0 + 256) * 16, 0, 1);

        // ---- B fragment from Kt[pixel][16] ----
        const _Float16* kp = kt + (p0 + (size_t)hi * 16) * 16 + lx;
        v16h bfrag;
        #pragma unroll
        for (int v = 0; v < 8; ++v) {
            bfrag[2 * v]     = kp[(2 * v) * 16];
            bfrag[2 * v + 1] = kp[(2 * v + 1) * 16];
        }
        // ---- A fragments from feat rows (contiguous f32) ----
        #pragma unroll
        for (int m = 0; m < 4; ++m) {
            const float* fp = feat + (((size_t)(b * 64 + m * 16 + lx)) << 16) + n0 + hi * 16;
            float4 a0 = *(const float4*)(fp);
            float4 a1 = *(const float4*)(fp + 4);
            float4 a2 = *(const float4*)(fp + 8);
            float4 a3 = *(const float4*)(fp + 12);
            v16h afrag;
            afrag[0]  = (_Float16)a0.x; afrag[1]  = (_Float16)a0.y;
            afrag[2]  = (_Float16)a0.z; afrag[3]  = (_Float16)a0.w;
            afrag[4]  = (_Float16)a1.x; afrag[5]  = (_Float16)a1.y;
            afrag[6]  = (_Float16)a1.z; afrag[7]  = (_Float16)a1.w;
            afrag[8]  = (_Float16)a2.x; afrag[9]  = (_Float16)a2.y;
            afrag[10] = (_Float16)a2.z; afrag[11] = (_Float16)a2.w;
            afrag[12] = (_Float16)a3.x; afrag[13] = (_Float16)a3.y;
            afrag[14] = (_Float16)a3.z; afrag[15] = (_Float16)a3.w;
            acc[m] = wmma_f16(afrag, bfrag, acc[m]);
        }
    }

    // flush: kft[b][o][16] += D
    #pragma unroll
    for (int m = 0; m < 4; ++m)
        #pragma unroll
        for (int r = 0; r < 8; ++r) {
            int o = m * 16 + r + hi * 8;
            atomicAdd(&kft[((size_t)(b * 64 + o)) * 16 + lx], acc[m][r]);
        }
}

// ================= K2c: fold v_w -> Vsum, matrix, Ksum+eps =================
__global__ void k2c_fold(const float* __restrict__ featsum, const float* __restrict__ ksum,
                         const float* __restrict__ kft,
                         const float* __restrict__ v_w, const float* __restrict__ v_b,
                         float* __restrict__ vsum, float* __restrict__ mat,
                         float* __restrict__ kseps)
{
    int b = blockIdx.x, c = threadIdx.x;
    float a = 0.0f;
    #pragma unroll 8
    for (int o = 0; o < COUT; ++o) a += v_w[c * COUT + o] * featsum[b * COUT + o];
    vsum[b * COUT + c] = a + (float)NSP * v_b[c];
    for (int q = 0; q < CQ; ++q) {
        float m = 0.0f;
        #pragma unroll 8
        for (int o = 0; o < COUT; ++o) m += v_w[c * COUT + o] * kft[(size_t)(b * 64 + o) * 16 + q];
        mat[(b * CQ + q) * COUT + c] = m + v_b[c] * ksum[b * CQ + q];
    }
    if (c < CQ) kseps[b * CQ + c] = ksum[b * CQ + c] + EPS_ATT;
}

// ================= K3: Q, tailor, output =================
__global__ __launch_bounds__(256) void k3_out(
    const float* __restrict__ feat,
    const float* __restrict__ q_w, const float* __restrict__ q_b,
    const float* __restrict__ vsum, const float* __restrict__ mat,
    const float* __restrict__ kseps, const float* __restrict__ gamma,
    float* __restrict__ out)
{
    __shared__ float qwl[CQ * COUT];
    __shared__ float matl[CQ * COUT];
    __shared__ float vsl[COUT];
    __shared__ float kse[CQ];

    const int tid = threadIdx.x;
    const int p = blockIdx.x * 256 + tid;
    const int b = p >> 16;
    const int n = p & (NSP - 1);

    for (int i = tid; i < CQ * COUT; i += 256) {
        qwl[i]  = q_w[i];
        matl[i] = mat[b * CQ * COUT + i];
    }
    if (tid < COUT) vsl[tid] = vsum[b * COUT + tid];
    if (tid < CQ)   kse[tid] = kseps[b * CQ + tid];
    __syncthreads();

    const size_t base = (((size_t)b) << 22) + n;
    float fr[COUT];
    #pragma unroll 8
    for (int o = 0; o < COUT; ++o) fr[o] = feat[base + ((size_t)o << 16)];

    // Q = l2norm(q_w @ feat + q_b)
    float qq[CQ];
    float s2 = 0.0f;
    #pragma unroll
    for (int q = 0; q < CQ; ++q) {
        float a = q_b[q];
        #pragma unroll 8
        for (int o = 0; o < COUT; ++o) a += qwl[q * COUT + o] * fr[o];
        qq[q] = a; s2 += a * a;
    }
    float rn = 1.0f / sqrtf(s2);
    float tdot = 0.0f;
    #pragma unroll
    for (int q = 0; q < CQ; ++q) { qq[q] *= rn; tdot += qq[q] * kse[q]; }
    const float gt = gamma[0] / ((float)NSP + tdot);   // gamma * tailor

    #pragma unroll 8
    for (int o = 0; o < COUT; ++o) {
        float ms = vsl[o];
        #pragma unroll
        for (int q = 0; q < CQ; ++q) ms += qq[q] * matl[q * COUT + o];
        out[base + ((size_t)o << 16)] = gt * ms + fr[o];
    }
}

// ================= launcher =================
extern "C" void kernel_launch(void* const* d_in, const int* in_sizes, int n_in,
                              void* d_out, int out_size, void* d_ws, size_t ws_size,
                              hipStream_t stream) {
    (void)in_sizes; (void)n_in; (void)out_size; (void)ws_size;
    const float* s5     = (const float*)d_in[0];
    const float* s4     = (const float*)d_in[1];
    const float* s3     = (const float*)d_in[2];
    const float* s2     = (const float*)d_in[3];
    const float* conv_w = (const float*)d_in[4];
    const float* bn_w   = (const float*)d_in[5];
    const float* bn_b   = (const float*)d_in[6];
    const float* q_w    = (const float*)d_in[7];
    const float* q_b    = (const float*)d_in[8];
    const float* k_w    = (const float*)d_in[9];
    const float* k_b    = (const float*)d_in[10];
    const float* v_w    = (const float*)d_in[11];
    const float* v_b    = (const float*)d_in[12];
    const float* gamma  = (const float*)d_in[13];

    float* ws    = (float*)d_ws;
    float* x     = ws + WS_X;                       // x / feat (in place)
    _Float16* kt = (_Float16*)(ws + WS_KT_F);
    float* st    = ws + WS_STATS;

    float* gsum    = st + ST_GSUM;
    float* gsumsq  = st + ST_GSUMSQ;
    float* featsum = st + ST_FEATSUM;
    float* ksum    = st + ST_KSUM;
    float* kft     = st + ST_KFT;
    float* scale   = st + ST_SCALE;
    float* shift   = st + ST_SHIFT;
    float* kseps   = st + ST_KSEPS;
    float* vsum    = st + ST_VSUM;
    float* mat     = st + ST_MAT;

    k0_zero<<<(ST_ZERO_N + 255) / 256, 256, 0, stream>>>(st);
    k1_conv<<<512, 256, 0, stream>>>(s5, s4, s3, s2, conv_w, x, gsum, gsumsq);
    k1b_bn<<<1, 64, 0, stream>>>(gsum, gsumsq, bn_w, bn_b, scale, shift);
    k2_feat<<<NPIX / 256, 256, 0, stream>>>(x, scale, shift, k_w, k_b, kt, featsum, ksum);
    k2b_kf<<<128, 256, 0, stream>>>(x, kt, kft);
    k2c_fold<<<NB, 64, 0, stream>>>(featsum, ksum, kft, v_w, v_b, vsum, mat, kseps);
    k3_out<<<NPIX / 256, 256, 0, stream>>>(x, q_w, q_b, vsum, mat, kseps, gamma, (float*)d_out);
}
